// BASE_RelPatchAttention2D_v2_56264071578234
// MI455X (gfx1250) — matrix operand order repaired and checked by reading, not compiled, not run on hardware
//
#include <hip/hip_runtime.h>

// ---------------------------------------------------------------------------
// RelPatchAttention2D for MI455X (gfx1250, wave32, WMMA).
// B=8, C=256, H=W=128, S0=4, S1=8 -> 256 patches of 16384 elements per batch.
// Phase 1: per-batch 256x256x16384 Gram GEMM in f16 WMMA (f32 accum) + norms.
// Phase 2: similarity ratio + weighted shrink over query-patch axis -> t[b,j].
// Phase 3: out = t[b, patch(j)] * v  (elementwise, float4).
// ---------------------------------------------------------------------------

#define NPATCH 256
#define CDIM   256
#define HDIM   128
#define WDIM   128
#define KTOT   16384          // 64 * 16 * 16 elements per patch
#define SMOOTHF 1e-5f

#define KSEG   16             // K-dimension split across workgroups
#define SEGLEN (KTOT / KSEG)  // 1024
#define KC     64             // K panel staged in LDS per iteration
#define BM     128            // output tile rows per workgroup
#define BN     128            // output tile cols per workgroup

typedef __attribute__((ext_vector_type(16))) _Float16 v16h;
typedef __attribute__((ext_vector_type(8)))  _Float16 v8h;
typedef __attribute__((ext_vector_type(8)))  float    v8f;

// workspace layout (float offsets)
#define G_OFF   0
#define QQ_OFF  (8 * NPATCH * NPATCH)            // 524288
#define KK_OFF  (QQ_OFF + 8 * NPATCH)            // +2048
#define T_OFF   (KK_OFF + 8 * NPATCH)            // +2048
#define ZERO_FLOATS (T_OFF)                      // zero G, qq, kk

__device__ __forceinline__ size_t patchElemOff(int b, int p, int e) {
  // patch p = (s0, s1a, s1b), element e = (cc, hh, ww)
  int s0 = p >> 6, s1a = (p >> 3) & 7, s1b = p & 7;
  int cc = e >> 8, hh = (e >> 4) & 15, ww = e & 15;
  int ch = (s0 << 6) + cc;
  int y  = (s1a << 4) + hh;
  int x  = (s1b << 4) + ww;
  return (size_t)(((b * CDIM + ch) * HDIM + y)) * WDIM + x;
}

__global__ __launch_bounds__(256)
void zero_ws_kernel(float* __restrict__ ws, int n) {
  int i = blockIdx.x * blockDim.x + threadIdx.x;
  if (i < n) ws[i] = 0.0f;
}

// Convert 16 consecutive patch elements (one memory row of 16 floats) to f16.
__device__ __forceinline__ v16h load16_cvt(const float* __restrict__ src) {
  const float4* s = (const float4*)src;
  v16h hv;
#pragma unroll
  for (int j = 0; j < 4; ++j) {
    float4 f = s[j];
    hv[4 * j + 0] = (_Float16)f.x;
    hv[4 * j + 1] = (_Float16)f.y;
    hv[4 * j + 2] = (_Float16)f.z;
    hv[4 * j + 3] = (_Float16)f.w;
  }
  return hv;
}

__global__ __launch_bounds__(256)
void gram_kernel(const float* __restrict__ Q, const float* __restrict__ Kin,
                 float* __restrict__ G, float* __restrict__ qq,
                 float* __restrict__ kkv) {
  __shared__ __align__(32) _Float16 aT[BM * KC];  // Q patches  [row][k]
  __shared__ __align__(32) _Float16 bT[BN * KC];  // K patches  [col][k]

  const int bid  = blockIdx.x;
  const int seg  = bid & (KSEG - 1);
  const int cb   = (bid >> 4) & 1;
  const int rb   = (bid >> 5) & 1;
  const int b    = bid >> 6;

  const int tid  = threadIdx.x;
  const int lane = tid & 31;
  const int wave = tid >> 5;        // 0..7, owns C rows [16*wave, 16*wave+16)

  const int rowBase = rb * BM;      // query patch base
  const int colBase = cb * BN;      // key patch base
  const int kBase   = seg * SEGLEN;

  v8f acc[8];
#pragma unroll
  for (int i = 0; i < 8; ++i) { v8f z = {}; acc[i] = z; }

  float qqAcc = 0.0f, kkAcc = 0.0f;

  // loader: thread t -> panel row lr = t>>1, 32 consecutive K (= 2 rows of 16)
  const int lr    = tid >> 1;
  const int lpart = tid & 1;

  for (int pk = 0; pk < SEGLEN; pk += KC) {
    const int k0 = kBase + pk;
    __syncthreads();  // previous panel fully consumed
    {
      const int e0 = k0 + lpart * 32;
      // A panel (Q)
      {
        const int p = rowBase + lr;
        size_t o0 = patchElemOff(b, p, e0);
        size_t o1 = patchElemOff(b, p, e0 + 16);
        *(v16h*)(&aT[lr * KC + lpart * 32])      = load16_cvt(Q + o0);
        *(v16h*)(&aT[lr * KC + lpart * 32 + 16]) = load16_cvt(Q + o1);
      }
      // B panel (K)
      {
        const int p = colBase + lr;
        size_t o0 = patchElemOff(b, p, e0);
        size_t o1 = patchElemOff(b, p, e0 + 16);
        *(v16h*)(&bT[lr * KC + lpart * 32])      = load16_cvt(Kin + o0);
        *(v16h*)(&bT[lr * KC + lpart * 32 + 16]) = load16_cvt(Kin + o1);
      }
    }
    __syncthreads();

    // Row/col self-norm partial sums (only where each K column is seen once).
    if (cb == 0 && tid < 128) {
#pragma unroll 8
      for (int j = 0; j < KC; ++j) {
        float v = (float)aT[tid * KC + j];
        qqAcc += v * v;
      }
    }
    if (rb == 0 && tid >= 128) {
      const int r = tid - 128;
#pragma unroll 8
      for (int j = 0; j < KC; ++j) {
        float v = (float)bT[r * KC + j];
        kkAcc += v * v;
      }
    }

    // WMMA compute: 2 K-steps of 32, 8 column tiles each.
    const int hi   = lane >> 4;
    const int l15  = lane & 15;
#pragma unroll
    for (int ks = 0; ks < KC; ks += 32) {
      // A fragment: halves 0..15 = A[r][kb..kb+15] (contiguous),
      // kb = ks + (lane>>4)*16, matching the B fragment.
      const int arow = wave * 16 + l15;
      const _Float16* ap = &aT[arow * KC + ks + hi * 16];
      v8h alo = *(const v8h*)ap;
      v8h ahi = *(const v8h*)(ap + 8);
      v16h afrag;
#pragma unroll
      for (int i = 0; i < 8; ++i) { afrag[i] = alo[i]; afrag[i + 8] = ahi[i]; }

#pragma unroll
      for (int ct = 0; ct < 8; ++ct) {
        // B fragment: lane holds col = lane&15, 16 contiguous K values
        // starting at ks + (lane>>4)*16 (stored col-major in bT).
        const int bcol = ct * 16 + l15;
        v16h bfrag = *(const v16h*)(&bT[bcol * KC + ks + hi * 16]);
        acc[ct] = __builtin_amdgcn_wmma_f32_16x16x32_f16(
            false, afrag, false, bfrag, (short)0, acc[ct], false, false);
      }
    }
  }

  // Accumulate partial Gram into global G (K-split => atomic add).
  float* Gb = G + (size_t)b * NPATCH * NPATCH;
  const int hi  = lane >> 4;
  const int l15 = lane & 15;
#pragma unroll
  for (int ct = 0; ct < 8; ++ct) {
    const int col = colBase + ct * 16 + l15;
#pragma unroll
    for (int v = 0; v < 8; ++v) {
      const int row = rowBase + wave * 16 + v + 8 * hi;
      atomicAdd(&Gb[row * NPATCH + col], acc[ct][v]);
    }
  }
  if (cb == 0 && tid < 128) atomicAdd(&qq[b * NPATCH + rowBase + tid], qqAcc);
  if (rb == 0 && tid >= 128)
    atomicAdd(&kkv[b * NPATCH + colBase + (tid - 128)], kkAcc);
}

__global__ __launch_bounds__(256)
void shrink_kernel(const float* __restrict__ G, const float* __restrict__ qq,
                   const float* __restrict__ kkv, const float* __restrict__ w,
                   const float* __restrict__ bias, float* __restrict__ t) {
  const int b = blockIdx.x;
  const int j = threadIdx.x;
  const float* Gb = G + (size_t)b * NPATCH * NPATCH;
  const float kj = kkv[b * NPATCH + j];
  float s = 0.0f;
  for (int i = 0; i < NPATCH; ++i) {
    float g = Gb[i * NPATCH + j];
    float d = qq[b * NPATCH + i] + kj - g + SMOOTHF;
    s += w[i] * (g + SMOOTHF) / d;
  }
  t[b * NPATCH + j] = s + bias[0];
}

__global__ __launch_bounds__(256)
void apply_kernel(const float* __restrict__ V, const float* __restrict__ t,
                  float* __restrict__ out) {
  const size_t i4 = (size_t)blockIdx.x * blockDim.x + threadIdx.x;
  const size_t g  = i4 * 4;           // float index; x%16 in {0,4,8,12} => one patch
  const int x  = (int)(g & 127);
  const int y  = (int)((g >> 7) & 127);
  const int ch = (int)((g >> 14) & 255);
  const int b  = (int)(g >> 22);
  const int n  = ((ch >> 6) << 6) | ((y >> 4) << 3) | (x >> 4);
  const float s = t[b * NPATCH + n];
  float4 v = ((const float4*)V)[i4];
  float4 o;
  o.x = s * v.x; o.y = s * v.y; o.z = s * v.z; o.w = s * v.w;
  ((float4*)out)[i4] = o;
}

extern "C" void kernel_launch(void* const* d_in, const int* in_sizes, int n_in,
                              void* d_out, int out_size, void* d_ws, size_t ws_size,
                              hipStream_t stream) {
  const float* q    = (const float*)d_in[0];
  const float* k    = (const float*)d_in[1];
  const float* v    = (const float*)d_in[2];
  const float* w    = (const float*)d_in[3];  // 256 weights
  const float* bias = (const float*)d_in[4];  // 1 element

  float* ws  = (float*)d_ws;
  float* G   = ws + G_OFF;
  float* qq  = ws + QQ_OFF;
  float* kkv = ws + KK_OFF;
  float* t   = ws + T_OFF;

  (void)in_sizes; (void)n_in; (void)out_size; (void)ws_size;

  zero_ws_kernel<<<(ZERO_FLOATS + 255) / 256, 256, 0, stream>>>(ws, ZERO_FLOATS);
  gram_kernel<<<8 * 2 * 2 * KSEG, 256, 0, stream>>>(q, k, G, qq, kkv);
  shrink_kernel<<<8, NPATCH, 0, stream>>>(G, qq, kkv, w, bias, t);
  apply_kernel<<<(8 * CDIM * HDIM * WDIM) / (4 * 256), 256, 0, stream>>>(
      v, t, (float*)d_out);
}
